// CrAKN_49813030699159
// MI455X (gfx1250) — matrix operand order repaired and checked by reading, not compiled, not run on hardware
//
#include <hip/hip_runtime.h>
#include <math.h>

typedef __attribute__((ext_vector_type(16))) __bf16 v16bf;
typedef __attribute__((ext_vector_type(8)))  float  v8f;

constexpr int NSEQ  = 512;
constexpr int NHEAD = 4;

// round-to-nearest-even f32 -> bf16 (bit level, avoids relying on __bf16 casts)
__device__ inline unsigned short f2bf(float f) {
  unsigned int u = __builtin_bit_cast(unsigned int, f);
  unsigned int r = (u + 0x7fffu + ((u >> 16) & 1u)) >> 16;
  return (unsigned short)r;
}

__device__ inline float mishf(float x) {
  float sp = log1pf(expf(x));   // softplus; inf-safe: tanh(inf)=1 -> mish->x
  return x * tanhf(sp);
}

// ---------------------------------------------------------------------------
// Pack Wb [512,128] and Wbo [128,512] (f32) into bf16 WMMA B-fragment order.
// wbp : [cb=32][ks=4][lane=32][16]   (B for GEMM1, K=128, Ncols=512)
// wbop: [cb2=8][ks2=16][lane=32][16] (B for GEMM2, K=512, Ncols=128)
// B 32x16 layout: lane col = lane&15, K = ks*32 + ((j<8) ? (lane>>4)*8+j : (lane>>4)*8+8+j)
// ---------------------------------------------------------------------------
__global__ __launch_bounds__(256) void pack_weights_kernel(
    const float* __restrict__ Wb, const float* __restrict__ Wbo,
    unsigned short* __restrict__ wbp, unsigned short* __restrict__ wbop) {
  int idx = blockIdx.x * 256 + threadIdx.x;
  if (idx >= 65536) return;
  {
    int j = idx & 15, ln = (idx >> 4) & 31, ks = (idx >> 9) & 3, cb = idx >> 11;
    int n = cb * 16 + (ln & 15);
    int k = ks * 32 + ((j < 8) ? ((ln >> 4) * 8 + j) : ((ln >> 4) * 8 + 8 + j));
    wbp[idx] = f2bf(Wb[n * 128 + k]);
  }
  {
    int j = idx & 15, ln = (idx >> 4) & 31, ks2 = (idx >> 9) & 15, cb2 = idx >> 13;
    int n = cb2 * 16 + (ln & 15);
    int k = ks2 * 32 + ((j < 8) ? ((ln >> 4) * 8 + j) : ((ln >> 4) * 8 + 8 + j));
    wbop[idx] = f2bf(Wbo[n * 512 + k]);
  }
}

// ---------------------------------------------------------------------------
// Fused bias path: per 32-row tile of the N*N flattened rows:
//   P = bias_tile @ Wb.T + bb   (GEMM1, bf16 WMMA, f32 accum)
//   diffs[row,h] = ||P[row, h*128:(h+1)*128]||_2   -> ws
//   bias_out = mish(P @ Wbo.T + bbo)               -> d_out
// P lives only in LDS (bf16, pre-swizzled as GEMM2 A-fragments).
// ---------------------------------------------------------------------------
__global__ __launch_bounds__(256) void bias_fused_kernel(
    const float* __restrict__ bias,
    const unsigned short* __restrict__ wbp,
    const unsigned short* __restrict__ wbop,
    const float* __restrict__ bb, const float* __restrict__ bbo,
    float* __restrict__ diffs, float* __restrict__ bias_out) {
  __shared__ __attribute__((aligned(32))) unsigned short atile[2 * 4 * 32 * 16];   // 8 KB
  __shared__ __attribute__((aligned(32))) unsigned short pbuf[2 * 16 * 32 * 16];   // 32 KB
  __shared__ float normacc[32 * 4];

  const int t    = threadIdx.x;
  const int lane = t & 31;
  const int wave = t >> 5;
  const int hh   = lane >> 4;
  const long rowbase = (long)blockIdx.x * 32;

  // Stage bias tile -> LDS bf16, pre-swizzled into A-fragment order.
  // A 16x32 layout: lane row = lane&15, K = ks*32 + ((j<8) ? h*8+j : h*8+8+j)
  #pragma unroll
  for (int s = 0; s < 16; ++s) {
    int slot = t * 16 + s;
    int j  = slot & 15;
    int ln = (slot >> 4) & 31;
    int ks = (slot >> 9) & 3;
    int rb = slot >> 11;
    int h  = ln >> 4;
    int m  = rb * 16 + (ln & 15);
    int k  = ks * 32 + ((j < 8) ? (h * 8 + j) : (h * 8 + 8 + j));
    atile[slot] = f2bf(bias[(rowbase + m) * 128 + k]);
  }
  __syncthreads();

  const int rb   = wave & 1;        // row-block (2 x 16 rows)
  const int head = wave >> 1;       // each wave owns one head's 128 columns
  const int cb0  = head * 8;        // 8 col-blocks of 16

  v16bf afrag[4];
  #pragma unroll
  for (int ks = 0; ks < 4; ++ks)
    afrag[ks] = *(const v16bf*)&atile[((rb * 4 + ks) * 32 + lane) * 16];

  float nacc[8];
  #pragma unroll
  for (int i = 0; i < 8; ++i) nacc[i] = 0.f;

  // ---- GEMM1: [32x128] x [128x512] -> P, this wave: rows rb*16.., cols head*128.. ----
  #pragma unroll
  for (int cbi = 0; cbi < 8; ++cbi) {
    int cb = cb0 + cbi;
    v8f acc = {0.f, 0.f, 0.f, 0.f, 0.f, 0.f, 0.f, 0.f};
    #pragma unroll
    for (int ks = 0; ks < 4; ++ks) {
      v16bf bfrag = *(const v16bf*)&wbp[((cb * 4 + ks) * 32 + lane) * 16];
      acc = __builtin_amdgcn_wmma_f32_16x16x32_bf16(false, afrag[ks], false, bfrag,
                                                    (short)0, acc, false, false);
    }
    int n    = cb * 16 + (lane & 15);      // global col 0..511 (k2 of GEMM2)
    float bbv = bb[n];
    int ks2 = n >> 5;
    int ko  = n & 31;
    int h2  = (ko >> 3) & 1;
    int j2  = (ko < h2 * 8 + 8) ? (ko - h2 * 8) : (ko - h2 * 8 - 8);
    #pragma unroll
    for (int i = 0; i < 8; ++i) {
      float x = acc[i] + bbv;              // C layout: M = i + 8*hh, N = lane&15
      nacc[i] += x * x;
      int mloc  = rb * 16 + i + hh * 8;
      int lane2 = h2 * 16 + (mloc & 15);
      pbuf[(((mloc >> 4) * 16 + ks2) * 32 + lane2) * 16 + j2] = f2bf(x);
    }
  }

  // Reduce sum-of-squares across the 16 lanes of each half (cols of the head).
  #pragma unroll
  for (int i = 0; i < 8; ++i) {
    float v = nacc[i];
    v += __shfl_xor(v, 1, 32);
    v += __shfl_xor(v, 2, 32);
    v += __shfl_xor(v, 4, 32);
    v += __shfl_xor(v, 8, 32);
    if ((lane & 15) == 0) {
      int mloc = rb * 16 + i + hh * 8;     // unique (mloc,head) per writer
      normacc[mloc * 4 + head] = v;
    }
  }
  __syncthreads();

  if (t < 128) {
    int mloc = t >> 2, h = t & 3;
    diffs[(rowbase + mloc) * 4 + h] = sqrtf(normacc[t]);
  }

  // ---- GEMM2: [32x512] x [512x128] -> bias_out tile ----
  #pragma unroll
  for (int c = 0; c < 2; ++c) {
    int cb2 = head * 2 + c;
    v8f acc = {0.f, 0.f, 0.f, 0.f, 0.f, 0.f, 0.f, 0.f};
    #pragma unroll
    for (int ks2 = 0; ks2 < 16; ++ks2) {
      v16bf a = *(const v16bf*)&pbuf[((rb * 16 + ks2) * 32 + lane) * 16];
      v16bf b = *(const v16bf*)&wbop[((cb2 * 16 + ks2) * 32 + lane) * 16];
      acc = __builtin_amdgcn_wmma_f32_16x16x32_bf16(false, a, false, b,
                                                    (short)0, acc, false, false);
    }
    int n   = cb2 * 16 + (lane & 15);
    float bo = bbo[n];
    #pragma unroll
    for (int i = 0; i < 8; ++i) {
      int mloc = rb * 16 + i + hh * 8;
      bias_out[(rowbase + mloc) * 128 + n] = mishf(acc[i] + bo);
    }
  }
}

// ---------------------------------------------------------------------------
// pq = q @ Wq.T + bq ; pk = k @ Wk.T + bk   (rows 0..511 -> pq, 512..1023 -> pk)
// ---------------------------------------------------------------------------
__global__ __launch_bounds__(256) void proj_kernel(
    const float* __restrict__ q, const float* __restrict__ k,
    const float* __restrict__ Wq, const float* __restrict__ bq,
    const float* __restrict__ Wk, const float* __restrict__ bk,
    float* __restrict__ pq, float* __restrict__ pk) {
  int row = blockIdx.x;
  const float *x, *W, *b; float* out;
  if (row < NSEQ) { x = q + row * 64; W = Wq; b = bq; out = pq + row * 512; }
  else { int r = row - NSEQ; x = k + r * 64; W = Wk; b = bk; out = pk + r * 512; }
  __shared__ float xs[64];
  if (threadIdx.x < 64) xs[threadIdx.x] = x[threadIdx.x];
  __syncthreads();
  for (int n = threadIdx.x; n < 512; n += 256) {
    float s = b[n];
    #pragma unroll 8
    for (int kk = 0; kk < 64; ++kk) s = fmaf(xs[kk], W[n * 64 + kk], s);
    out[n] = s;
  }
}

// ---------------------------------------------------------------------------
// Attention: logits = pq.pk/sqrt(128) + diffs; softmax over k; vals = attn@v;
// vals_mean = mean over heads. One block per q row.
// ---------------------------------------------------------------------------
__device__ inline float blk_max(float x, float* red) {
  int t = threadIdx.x;
  red[t] = x; __syncthreads();
  for (int s = 128; s > 0; s >>= 1) { if (t < s) red[t] = fmaxf(red[t], red[t + s]); __syncthreads(); }
  float r = red[0]; __syncthreads();
  return r;
}
__device__ inline float blk_sum(float x, float* red) {
  int t = threadIdx.x;
  red[t] = x; __syncthreads();
  for (int s = 128; s > 0; s >>= 1) { if (t < s) red[t] += red[t + s]; __syncthreads(); }
  float r = red[0]; __syncthreads();
  return r;
}

__global__ __launch_bounds__(256) void attn_kernel(
    const float* __restrict__ pq, const float* __restrict__ pk,
    const float* __restrict__ v, const float* __restrict__ diffs,
    float* __restrict__ vals_mean) {
  __shared__ float pqrow[512];
  __shared__ float red[256];
  int t = threadIdx.x;
  int qr = blockIdx.x;
  pqrow[t]       = pq[qr * 512 + t];
  pqrow[t + 256] = pq[qr * 512 + t + 256];
  __syncthreads();
  float vh = 0.f;
  for (int h = 0; h < NHEAD; ++h) {
    float lg0, lg1;
    {
      const float* pkr = pk + t * 512 + h * 128;
      float d0 = 0.f;
      #pragma unroll 8
      for (int d = 0; d < 128; ++d) d0 = fmaf(pqrow[h * 128 + d], pkr[d], d0);
      lg0 = d0 * 0.08838834764831845f + diffs[((long)qr * 512 + t) * 4 + h];
    }
    {
      const float* pkr = pk + (t + 256) * 512 + h * 128;
      float d0 = 0.f;
      #pragma unroll 8
      for (int d = 0; d < 128; ++d) d0 = fmaf(pqrow[h * 128 + d], pkr[d], d0);
      lg1 = d0 * 0.08838834764831845f + diffs[((long)qr * 512 + t + 256) * 4 + h];
    }
    float mx = blk_max(fmaxf(lg0, lg1), red);
    float e0 = expf(lg0 - mx), e1 = expf(lg1 - mx);
    float Z  = blk_sum(e0 + e1, red);
    float wv = blk_sum(e0 * v[t] + e1 * v[t + 256], red);
    vh += wv / Z;
  }
  if (t == 0) vals_mean[qr] = vh * 0.25f;
}

// ---------------------------------------------------------------------------
// q_new / k_new: (d,h)-major flatten of proj, @ W{q,k}o.T + b, mish, residual,
// layernorm over 64. One block (64 threads) per row; rows >=512 -> k path.
// ---------------------------------------------------------------------------
__global__ __launch_bounds__(64) void qknew_kernel(
    const float* __restrict__ q, const float* __restrict__ k,
    const float* __restrict__ pq, const float* __restrict__ pk,
    const float* __restrict__ Wqo, const float* __restrict__ bqo,
    const float* __restrict__ Wko, const float* __restrict__ bko,
    const float* __restrict__ qg, const float* __restrict__ qb,
    const float* __restrict__ kg, const float* __restrict__ kb,
    float* __restrict__ q_new, float* __restrict__ k_new) {
  int row = blockIdx.x, t = threadIdx.x;
  const float *p, *x, *W, *b, *g, *be; float* out;
  if (row < NSEQ) { p = pq + row * 512; x = q + row * 64; W = Wqo; b = bqo; g = qg; be = qb; out = q_new + row * 64; }
  else { int r = row - NSEQ; p = pk + r * 512; x = k + r * 64; W = Wko; b = bko; g = kg; be = kb; out = k_new + r * 64; }
  __shared__ float ps[512];
  __shared__ float red[64];
  for (int i = t; i < 512; i += 64) ps[i] = p[i];
  __syncthreads();
  float s = b[t];
  #pragma unroll 4
  for (int i = 0; i < 512; ++i) {
    int h = i & 3, d = i >> 2;            // pq_flat[m, d*H+h] = pq[m, h, d]
    s = fmaf(ps[h * 128 + d], W[t * 512 + i], s);
  }
  float val = x[t] + mishf(s);
  red[t] = val; __syncthreads();
  for (int st = 32; st > 0; st >>= 1) { if (t < st) red[t] += red[t + st]; __syncthreads(); }
  float mean = red[0] * (1.f / 64.f); __syncthreads();
  float dv = val - mean;
  red[t] = dv * dv; __syncthreads();
  for (int st = 32; st > 0; st >>= 1) { if (t < st) red[t] += red[t + st]; __syncthreads(); }
  float var = red[0] * (1.f / 64.f); __syncthreads();
  out[t] = dv * rsqrtf(var + 1e-5f) * g[t] + be[t];
}

// ---------------------------------------------------------------------------
extern "C" void kernel_launch(void* const* d_in, const int* in_sizes, int n_in,
                              void* d_out, int out_size, void* d_ws, size_t ws_size,
                              hipStream_t stream) {
  (void)in_sizes; (void)n_in; (void)out_size; (void)ws_size;
  const float* q    = (const float*)d_in[0];
  const float* k    = (const float*)d_in[1];
  const float* v    = (const float*)d_in[2];
  const float* bias = (const float*)d_in[3];
  const float* Wq   = (const float*)d_in[4];
  const float* bq   = (const float*)d_in[5];
  const float* Wk   = (const float*)d_in[6];
  const float* bk   = (const float*)d_in[7];
  const float* Wqo  = (const float*)d_in[8];
  const float* bqo  = (const float*)d_in[9];
  const float* Wko  = (const float*)d_in[10];
  const float* bko  = (const float*)d_in[11];
  const float* qg   = (const float*)d_in[12];
  const float* qb   = (const float*)d_in[13];
  const float* kg   = (const float*)d_in[14];
  const float* kb   = (const float*)d_in[15];
  const float* Wb   = (const float*)d_in[16];
  const float* bb   = (const float*)d_in[17];
  const float* Wbo  = (const float*)d_in[18];
  const float* bbo  = (const float*)d_in[19];

  char* ws = (char*)d_ws;
  unsigned short* wbp  = (unsigned short*)(ws + 0);        // 128 KB
  unsigned short* wbop = (unsigned short*)(ws + 131072);   // 128 KB
  float* diffs = (float*)(ws + 262144);                    // 4 MB  [N*N][H]
  float* pq    = (float*)(ws + 4456448);                   // 1 MB
  float* pk    = (float*)(ws + 5505024);                   // 1 MB

  float* out      = (float*)d_out;
  float* q_new    = out;            // 512*64
  float* k_new    = out + 32768;    // 512*64
  float* vmean    = out + 65536;    // 512
  float* bias_out = out + 66048;    // 512*512*128

  pack_weights_kernel<<<256, 256, 0, stream>>>(Wb, Wbo, wbp, wbop);
  proj_kernel<<<1024, 256, 0, stream>>>(q, k, Wq, bq, Wk, bk, pq, pk);
  bias_fused_kernel<<<8192, 256, 0, stream>>>(bias, wbp, wbop, bb, bbo, diffs, bias_out);
  attn_kernel<<<512, 256, 0, stream>>>(pq, pk, v, diffs, vmean);
  qknew_kernel<<<1024, 64, 0, stream>>>(q, k, pq, pk, Wqo, bqo, Wko, bko,
                                        qg, qb, kg, kb, q_new, k_new);
}
